// TabularAugmentedModel_23270132810350
// MI455X (gfx1250) — matrix operand order repaired and checked by reading, not compiled, not run on hardware
//
#include <hip/hip_runtime.h>
#include <math.h>

// HMM forward filter, linear-space reformulation:
//   log-sum-exp(log_q + log_Tr)  ==  log(q @ softmax(Tr))
// The scan runs on normalized probabilities; the 32x32 predict contraction is
// a real matmul -> v_wmma_f32_16x16x32_f16.
//
// Per wave (wave32), tile of 16 episodes:
//   D[M=s'(2 tiles of 16), N=b(16)] += A[M,K=s] * B[K,N]
//   A = Tr[a]^T fp16 A-fragments, resident in VGPRs (16 frags, loaded once)
//   B = (a_b==a ? w_hat : 0) fp16 B-fragment, rebuilt each step
//   8 actions x 2 M-tiles = 16 WMMAs per step, straight-line so the per-action
//   dword masks (v_dual_cndmask_b32) co-execute with the in-flight XDL WMMAs.
// State q is carried directly in the C/D fragment register layout.

typedef __attribute__((ext_vector_type(16))) _Float16 v16h;
typedef __attribute__((ext_vector_type(8)))  float    v8f;
typedef __attribute__((ext_vector_type(8)))  unsigned v8u;

#define S_N 32
#define O_N 64
#define A_N 8
#define R_N 8
#define B_N 1024
#define T_N 128

__global__ __launch_bounds__(32) void hmm_forward_wmma(
    const float* __restrict__ regime,   // [B]
    const int*   __restrict__ obs,      // [T+1, B]
    const int*   __restrict__ rew,      // [T+1, B]
    const float* __restrict__ don,      // [T+1, B]
    const int*   __restrict__ act,      // [T,   B]
    const float* __restrict__ p_s,      // [S]
    const float* __restrict__ p_ssa,    // [S, A, S]
    const float* __restrict__ p_os,     // [S, O]
    const float* __restrict__ p_rs,     // [S, R]
    const float* __restrict__ p_as,     // [S, A]
    float*       __restrict__ out)      // [B]
{
    __shared__ __align__(16) float s_Ot[O_N * S_N];        // prob, transposed [o][s]
    __shared__ __align__(16) float s_Rt[R_N * S_N];        // prob, transposed [r][s]
    __shared__ __align__(16) float s_At[A_N * S_N];        // prob, transposed [a][s]
    __shared__ __align__(16) float s_Tr[A_N * S_N * S_N];  // prob, [a][s][s']
    __shared__ float s_pi[S_N];

    const int lane = threadIdx.x;   // 0..31
    const int lh   = lane >> 4;     // wave half (0/1)
    const int ln   = lane & 15;     // batch column within 16-episode tile

    // ---------------- setup: softmax tables (linear space) ----------------
    {   // initial distribution pi[s]
        float m = -INFINITY;
        for (int i = 0; i < S_N; ++i) m = fmaxf(m, p_s[i]);
        float sum = 0.f;
        for (int i = 0; i < S_N; ++i) sum += expf(p_s[i] - m);
        s_pi[lane] = expf(p_s[lane] - m) / sum;
    }
    {   // emission O: softmax over O per state row, stored transposed
        const int s = lane;
        float m = -INFINITY;
        for (int o = 0; o < O_N; ++o) m = fmaxf(m, p_os[s * O_N + o]);
        float sum = 0.f;
        for (int o = 0; o < O_N; ++o) sum += expf(p_os[s * O_N + o] - m);
        const float inv = 1.f / sum;
        for (int o = 0; o < O_N; ++o) s_Ot[o * S_N + s] = expf(p_os[s * O_N + o] - m) * inv;
    }
    {   // emission R
        const int s = lane;
        float m = -INFINITY;
        for (int r = 0; r < R_N; ++r) m = fmaxf(m, p_rs[s * R_N + r]);
        float sum = 0.f;
        for (int r = 0; r < R_N; ++r) sum += expf(p_rs[s * R_N + r] - m);
        const float inv = 1.f / sum;
        for (int r = 0; r < R_N; ++r) s_Rt[r * S_N + s] = expf(p_rs[s * R_N + r] - m) * inv;
    }
    {   // action model A
        const int s = lane;
        float m = -INFINITY;
        for (int a = 0; a < A_N; ++a) m = fmaxf(m, p_as[s * A_N + a]);
        float sum = 0.f;
        for (int a = 0; a < A_N; ++a) sum += expf(p_as[s * A_N + a] - m);
        const float inv = 1.f / sum;
        for (int a = 0; a < A_N; ++a) s_At[a * S_N + s] = expf(p_as[s * A_N + a] - m) * inv;
    }
    // transition: softmax over s' per (s,a) row; store as Tr[a][s][s']
    for (int row = lane; row < S_N * A_N; row += 32) {
        const int s = row >> 3, a = row & 7;
        const float* src = &p_ssa[(s * A_N + a) * S_N];
        float m = -INFINITY;
        for (int sp = 0; sp < S_N; ++sp) m = fmaxf(m, src[sp]);
        float sum = 0.f;
        for (int sp = 0; sp < S_N; ++sp) sum += expf(src[sp] - m);
        const float inv = 1.f / sum;
        for (int sp = 0; sp < S_N; ++sp)
            s_Tr[a * (S_N * S_N) + s * S_N + sp] = expf(src[sp] - m) * inv;
    }
    __syncthreads();

    // ------------- load Tr^T A-fragments (fp16), resident in VGPRs -------------
    // A-fragment k order used here (contiguous, same as B): lane half selects K offset 0/16,
    // element e -> K = e + 16*lh ; M = ln (+16 for tile 1).
    v16h afr[A_N][2];
#pragma unroll
    for (int a = 0; a < A_N; ++a)
#pragma unroll
        for (int mt = 0; mt < 2; ++mt)
#pragma unroll
            for (int e = 0; e < 16; ++e) {
                const int K  = e + 16 * lh;                         // s
                const int M  = mt * 16 + ln;                        // s'
                afr[a][mt][e] = (_Float16)s_Tr[a * (S_N * S_N) + K * S_N + M];
            }

    // ------------- carried state in C/D fragment layout -------------
    // q[i]   (i<8):  state s = 8*lh + i        (D tile 0)
    // q[8+i] (i<8):  state s = 16 + 8*lh + i   (D tile 1)
    float q[16];
#pragma unroll
    for (int i = 0; i < 8; ++i) { q[i] = s_pi[8 * lh + i]; q[8 + i] = s_pi[16 + 8 * lh + i]; }

    const int   b        = blockIdx.x * 16 + ln;   // duplicated in both halves
    const float regv     = regime[b];
    float       lp       = 0.f;
    float       was_done = 0.f;

    // software-pipelined per-step scalars (issue t+1 loads before t's math)
    int   o_c = obs[b];
    int   r_c = rew[b];
    float d_c = don[b];
    int   a_c = act[b];

#pragma unroll 1
    for (int t = 0; t < T_N; ++t) {
        const int   o = o_c;
        const int   r = r_c;
        const float d = d_c;
        const int   a = a_c;
        {   // prefetch next step's scalars (row T of obs/rew/don is valid & used by tail)
            const int tn = t + 1;
            const int ta = (tn < T_N) ? tn : (T_N - 1);
            o_c = obs[tn * B_N + b];
            r_c = rew[tn * B_N + b];
            d_c = don[tn * B_N + b];
            a_c = act[ta * B_N + b];
        }

        const float done = fmaxf(was_done, d);
        const bool  aone = (done == 1.0f) || (regv == 1.0f);

        // gather emission rows as 4x ds_load_b128 per table
        float Of[16], Rf[16], Af[16];
        {
            const float4* pO = reinterpret_cast<const float4*>(s_Ot + o * S_N);
            const float4* pR = reinterpret_cast<const float4*>(s_Rt + r * S_N);
            const float4* pA = reinterpret_cast<const float4*>(s_At + a * S_N);
            const int iA = 2 * lh;        // run s = 8*lh   .. +7
            const int iB = 4 + 2 * lh;    // run s = 16+8*lh.. +7
            *(float4*)&Of[0]  = pO[iA];     *(float4*)&Of[4]  = pO[iA + 1];
            *(float4*)&Of[8]  = pO[iB];     *(float4*)&Of[12] = pO[iB + 1];
            *(float4*)&Rf[0]  = pR[iA];     *(float4*)&Rf[4]  = pR[iA + 1];
            *(float4*)&Rf[8]  = pR[iB];     *(float4*)&Rf[12] = pR[iB + 1];
            *(float4*)&Af[0]  = pA[iA];     *(float4*)&Af[4]  = pA[iA + 1];
            *(float4*)&Af[8]  = pA[iB];     *(float4*)&Af[12] = pA[iB + 1];
        }

        // filter: w[b,s] = q[b,s] * O[o,s] * R[r,s] * (A[a,s] or 1)
        float w[16];
#pragma unroll
        for (int i = 0; i < 16; ++i)
            w[i] = q[i] * Of[i] * Rf[i] * (aone ? 1.f : Af[i]);

        // normalizer over all 32 states: in-lane sum + half-wave exchange
        float nrm = 0.f;
#pragma unroll
        for (int i = 0; i < 16; ++i) nrm += w[i];
        nrm += __shfl_xor(nrm, 16);

        // log-likelihood accumulation (frozen after done; -inf sticky)
        const float add = (was_done == 0.f) ? logf(nrm) : 0.f;
        if (lp != -INFINITY) lp += add;

        const float inv = 1.f / nrm;
#pragma unroll
        for (int i = 0; i < 16; ++i) w[i] *= inv;

        // repack w_hat into B-fragment layout (K = 16*lh + e, N = ln):
        // lanes 0-15 need s=0..15, lanes 16-31 need s=16..31 -> cross-half swap
        float t1[8], t2[8];
#pragma unroll
        for (int e = 0; e < 8; ++e) { t1[e] = __shfl_xor(w[e], 16); t2[e] = __shfl_xor(w[8 + e], 16); }
        v16h bf;
#pragma unroll
        for (int e = 0; e < 8; ++e) {
            bf[e]     = (_Float16)(lh ? t2[e] : w[e]);
            bf[8 + e] = (_Float16)(lh ? w[8 + e] : t1[e]);
        }
        const v8u bu = __builtin_bit_cast(v8u, bf);

        // predict: D = sum_a Tr[a]^T x (mask_a .* w_hat), straight-line so the
        // next action's dword masks fill the current WMMAs' co-exec slots.
        v8f c0 = {}, c1 = {};
#pragma unroll
        for (int a2 = 0; a2 < A_N; ++a2) {
            const bool sel = (a == a2);
            v8u mu;
#pragma unroll
            for (int j = 0; j < 8; ++j) mu[j] = sel ? bu[j] : 0u;
            const v16h mb = __builtin_bit_cast(v16h, mu);
            c0 = __builtin_amdgcn_wmma_f32_16x16x32_f16(false, afr[a2][0], false, mb,
                                                        (short)0, c0, false, false);
            c1 = __builtin_amdgcn_wmma_f32_16x16x32_f16(false, afr[a2][1], false, mb,
                                                        (short)0, c1, false, false);
        }
#pragma unroll
        for (int i = 0; i < 8; ++i) { q[i] = c0[i]; q[8 + i] = c1[i]; }

        was_done = done;
    }

    // final step t = T: emissions only, no action, no transition.
    // o_c / r_c already hold row T from the pipelined loads.
    {
        float Of[16], Rf[16];
        const float4* pO = reinterpret_cast<const float4*>(s_Ot + o_c * S_N);
        const float4* pR = reinterpret_cast<const float4*>(s_Rt + r_c * S_N);
        const int iA = 2 * lh, iB = 4 + 2 * lh;
        *(float4*)&Of[0] = pO[iA];  *(float4*)&Of[4]  = pO[iA + 1];
        *(float4*)&Of[8] = pO[iB];  *(float4*)&Of[12] = pO[iB + 1];
        *(float4*)&Rf[0] = pR[iA];  *(float4*)&Rf[4]  = pR[iA + 1];
        *(float4*)&Rf[8] = pR[iB];  *(float4*)&Rf[12] = pR[iB + 1];

        float nrm = 0.f;
#pragma unroll
        for (int i = 0; i < 16; ++i) nrm += q[i] * Of[i] * Rf[i];
        nrm += __shfl_xor(nrm, 16);
        const float add = (was_done == 0.f) ? logf(nrm) : 0.f;
        if (lp != -INFINITY) lp += add;
    }

    if (lane < 16) out[b] = lp;
}

extern "C" void kernel_launch(void* const* d_in, const int* in_sizes, int n_in,
                              void* d_out, int out_size, void* d_ws, size_t ws_size,
                              hipStream_t stream) {
    const float* regime = (const float*)d_in[0];
    const int*   obs    = (const int*)  d_in[1];
    const int*   rew    = (const int*)  d_in[2];
    const float* don    = (const float*)d_in[3];
    const int*   act    = (const int*)  d_in[4];
    const float* p_s    = (const float*)d_in[5];
    const float* p_ssa  = (const float*)d_in[6];
    const float* p_os   = (const float*)d_in[7];
    const float* p_rs   = (const float*)d_in[8];
    const float* p_as   = (const float*)d_in[9];
    float* out = (float*)d_out;

    // 1024 episodes / 16 per wave, 1 wave (32 threads) per block
    hmm_forward_wmma<<<B_N / 16, 32, 0, stream>>>(
        regime, obs, rew, don, act, p_s, p_ssa, p_os, p_rs, p_as, out);
}
